// AEVComputer_44710609551814
// MI455X (gfx1250) — matrix operand order repaired and checked by reading, not compiled, not run on hardware
//
#include <hip/hip_runtime.h>
#include <hip/hip_bf16.h>
#include <math.h>

// ---------------------------------------------------------------------------
// AEV (ANI) computation for N=8 molecules x A=64 atoms, NUM_SPECIES=4.
// Radial:  out[i, s*16+t]        = sum_j  rad[i,j,t]   * onehot_s(species[j])
// Angular: out[i, 64+p*32+t]     = 0.5*sum_jk term[i,j,k,t]*onehot_p(pair(j,k))
// Both one-hot contractions are executed with v_wmma_f32_16x16x32_f16:
//   A = f16 tile of term/rad values (t rows x 32 jk cols), B = exact one-hot.
// One wave32 per atom; 8 waves (256 threads) per block; 64 blocks.
// ---------------------------------------------------------------------------

typedef __attribute__((ext_vector_type(16))) _Float16 v16h;
typedef __attribute__((ext_vector_type(8)))  float    v8f;

#define WAVES      8
#define AT_STRIDE  40           // halves per tile row (80B: 16B-aligned rows, bank-spread)
#define RCR_F      5.2f
#define RCA_F      3.5f
#define PI_F       3.14159265358979323846f

union AFrag { uint4 q[2]; v16h v; };
union BFrag { _Float16 h[16]; v16h v; };
union PBuf  { int4 q[4]; int s[16]; };

__device__ __forceinline__ float fcut(float d, float rc, float pi_over_rc) {
    return (d <= rc) ? (0.5f * __cosf(d * pi_over_rc) + 0.5f) : 0.0f;
}

// 16-bit A-matrix 16x32 fragment (ISA 7.12.2):
// lanes 0-15:  row M=lane,    halves 0-7 -> K=0..7,   halves 8-15 -> K=16..23
// lanes 16-31: row M=lane-16, halves 0-7 -> K=8..15,  halves 8-15 -> K=24..31
__device__ __forceinline__ v16h load_a_frag(const _Float16* tile, int lane, int row_off) {
    int row = (lane & 15) + row_off;
    int kb  = (lane < 16) ? 0 : 8;
    const _Float16* p = tile + row * AT_STRIDE + kb;
    AFrag f;
    f.q[0] = *reinterpret_cast<const uint4*>(p);        // K quad 1 (16B aligned)
    f.q[1] = *reinterpret_cast<const uint4*>(p + 16);   // K quad 2
    return f.v;
}

// 16-bit B-matrix 32x16 one-hot fragment:
// lane holds column N=lane&15; lanes 0-15 cover K=0..7,16..23, lanes 16-31 K=8..15,24..31;
// halves 0-7 -> K=kb..kb+7, halves 8-15 -> K=kb+16..kb+23.
__device__ __forceinline__ v16h make_b_frag(const int* pvec, int lane) {
    int col = lane & 15;
    int kb  = (lane < 16) ? 0 : 8;
    PBuf pb;
    const int4* q = reinterpret_cast<const int4*>(pvec + kb);
    pb.q[0] = q[0]; pb.q[1] = q[1]; pb.q[2] = q[4]; pb.q[3] = q[5];
    BFrag b;
#pragma unroll
    for (int e = 0; e < 16; ++e)
        b.h[e] = (pb.s[e] == col) ? (_Float16)1.0f : (_Float16)0.0f;
    return b.v;
}

__global__ __launch_bounds__(256, 1)
void aev_wmma_kernel(const int* __restrict__ species,
                     const float* __restrict__ coords,
                     const float* __restrict__ EtaR,
                     const float* __restrict__ ShfR,
                     const float* __restrict__ EtaA,
                     const float* __restrict__ Zeta,
                     const float* __restrict__ ShfA,
                     const float* __restrict__ ShfZ,
                     float* __restrict__ out)
{
    __shared__ float sx[64], sy[64], sz[64];
    __shared__ int   ssp[64];
    __shared__ __align__(16) _Float16 atile[WAVES][32 * AT_STRIDE];
    __shared__ __align__(16) int      ptile[WAVES][32];

    const int tid = threadIdx.x;
    const int n   = blockIdx.x >> 3;                 // molecule (8 blocks per molecule)
    if (tid < 64) {
        int ai = n * 64 + tid;
        ssp[tid] = species[ai];
        sx[tid]  = coords[ai * 3 + 0];
        sy[tid]  = coords[ai * 3 + 1];
        sz[tid]  = coords[ai * 3 + 2];
    }
    __syncthreads();

    const int wv   = tid >> 5;
    const int lane = tid & 31;
    const int i    = (blockIdx.x & 7) * 8 + wv;      // atom within molecule (0..63)

    const float xi = sx[i], yi = sy[i], zi = sz[i];
    const float etaR = EtaR[0], etaA = EtaA[0], zeta = Zeta[0];

    float shfr[16];
#pragma unroll
    for (int t = 0; t < 16; ++t) shfr[t] = ShfR[t];
    float shfa[4];
#pragma unroll
    for (int a = 0; a < 4; ++a) shfa[a] = ShfA[a];
    float czv[8], szv[8];
#pragma unroll
    for (int z = 0; z < 8; ++z) { float s = ShfZ[z]; czv[z] = __cosf(s); szv[z] = __sinf(s); }

    const float pi_rcr = PI_F / RCR_F;
    const float pi_rca = PI_F / RCA_F;

    // Per-lane precompute for the two k-columns this lane owns (k = lane, lane+32).
    float kx[2], ky[2], kz[2], kd[2], kfc[2], kinv[2], kfcr[2];
    int   ksp[2]; bool kval[2], krm[2];
#pragma unroll
    for (int h = 0; h < 2; ++h) {
        int k = lane + 32 * h;
        float dx = sx[k] - xi, dy = sy[k] - yi, dz = sz[k] - zi;
        float d  = sqrtf(dx * dx + dy * dy + dz * dz);
        kx[h] = dx; ky[h] = dy; kz[h] = dz; kd[h] = d;
        bool va = (k != i) && (d <= RCA_F);
        kval[h] = va;
        kinv[h] = va ? (1.0f / d) : 0.0f;
        kfc[h]  = fcut(d, RCA_F, pi_rca);
        kfcr[h] = fcut(d, RCR_F, pi_rcr);
        krm[h]  = (k != i) && (d <= RCR_F);
        ksp[h]  = ssp[k];
    }

    _Float16* mytile = atile[wv];
    int*      mypt   = ptile[wv];

    // ------------------------- Radial: 2 WMMA chunks -------------------------
    v8f c_rad = {};
#pragma unroll
    for (int h = 0; h < 2; ++h) {
        float d = kd[h];
        float m = krm[h] ? (0.25f * kfcr[h]) : 0.0f;
#pragma unroll
        for (int t = 0; t < 16; ++t) {
            float u = d - shfr[t];
            float v = m * __expf(-etaR * u * u);
            mytile[t * AT_STRIDE + lane] = (_Float16)v;
        }
        mypt[lane] = ksp[h];
        asm volatile("s_wait_dscnt 0" ::: "memory");   // same-wave LDS RAW (DS in-order)
        v16h a = load_a_frag(mytile, lane, 0);
        v16h b = make_b_frag(mypt, lane);
        c_rad = __builtin_amdgcn_wmma_f32_16x16x32_f16(false, a, false, b,
                                                       (short)0, c_rad, false, false);
        asm volatile("" ::: "memory");                 // keep next stores after these loads
    }

    // ---------------------- Angular: 128 WMMA-pair chunks --------------------
    v8f c_lo = {};
    v8f c_hi = {};
    for (int c = 0; c < 128; ++c) {
        const int j = c >> 1;
        const int h = c & 1;
        const int k = lane + 32 * h;

        float dxj = sx[j] - xi, dyj = sy[j] - yi, dzj = sz[j] - zi;
        float dj  = sqrtf(dxj * dxj + dyj * dyj + dzj * dzj);
        bool  vj  = (j != i) && (dj <= RCA_F);
        float invj = vj ? (1.0f / dj) : 0.0f;
        float fcj  = fcut(dj, RCA_F, pi_rca);

        float dot    = dxj * kx[h] + dyj * ky[h] + dzj * kz[h];
        float cosang = 0.95f * dot * invj * kinv[h];
        float sinang = sqrtf(fmaxf(0.0f, 1.0f - cosang * cosang));
        bool  vv     = vj && kval[h] && (j != k);
        float factor = vv ? (2.0f * fcj * kfc[h]) : 0.0f;
        float dh     = 0.5f * (dj + kd[h]);

        float e2[4];
#pragma unroll
        for (int a = 0; a < 4; ++a) {
            float u = dh - shfa[a];
            e2[a] = factor * __expf(-etaA * u * u);
        }
#pragma unroll
        for (int z = 0; z < 8; ++z) {
            // cos(theta - ShfZ) = cos*cz + sin*sz ; f1 = ((1+cos)/2)^zeta
            float f1b = fmaxf(0.0f, 0.5f * (1.0f + cosang * czv[z] + sinang * szv[z]));
            float f1  = __powf(f1b, zeta);   // log(0)->-inf -> exp->0
#pragma unroll
            for (int a = 0; a < 4; ++a)
                mytile[(a * 8 + z) * AT_STRIDE + lane] = (_Float16)(e2[a] * f1);
        }

        // pair index p = triu[s_j][s_k] for S=4: p = a*S - a(a-1)/2 + (b-a)
        int sj = ssp[j], sk = ksp[h];
        int pa = min(sj, sk), pbb = max(sj, sk);
        mypt[lane] = pa * 4 - ((pa * (pa - 1)) >> 1) + (pbb - pa);

        asm volatile("s_wait_dscnt 0" ::: "memory");
        v16h alo = load_a_frag(mytile, lane, 0);
        v16h ahi = load_a_frag(mytile, lane, 16);
        v16h bb  = make_b_frag(mypt, lane);
        c_lo = __builtin_amdgcn_wmma_f32_16x16x32_f16(false, alo, false, bb,
                                                      (short)0, c_lo, false, false);
        c_hi = __builtin_amdgcn_wmma_f32_16x16x32_f16(false, ahi, false, bb,
                                                      (short)0, c_hi, false, false);
        asm volatile("" ::: "memory");
    }

    // ------------------------------ Store out -------------------------------
    // out = [species as f32 (512)] ++ [aev (512 x 384)]
    float* aev = out + 512 + (n * 64 + i) * 384;
    const int col = lane & 15;
    const int bt  = (lane < 16) ? 0 : 8;             // C/D layout: VGPR r -> M = bt + r

    if (col < 4) {                                    // radial block: [s*16 + t]
#pragma unroll
        for (int r = 0; r < 8; ++r)
            aev[col * 16 + bt + r] = c_rad[r];
    }
    if (col < 10) {                                   // angular block: [64 + p*32 + t]
#pragma unroll
        for (int r = 0; r < 8; ++r) {
            aev[64 + col * 32 + bt + r]       = 0.5f * c_lo[r];
            aev[64 + col * 32 + 16 + bt + r]  = 0.5f * c_hi[r];
        }
    }
    if (lane == 0)
        out[n * 64 + i] = (float)ssp[i];
}

extern "C" void kernel_launch(void* const* d_in, const int* in_sizes, int n_in,
                              void* d_out, int out_size, void* d_ws, size_t ws_size,
                              hipStream_t stream) {
    (void)n_in; (void)out_size; (void)d_ws; (void)ws_size;
    const int*   species = (const int*)  d_in[0];  // (8,64)
    const float* coords  = (const float*)d_in[1];  // (8,64,3)
    const float* etaR    = (const float*)d_in[2];
    const float* shfR    = (const float*)d_in[3];  // 16
    const float* etaA    = (const float*)d_in[4];
    const float* zeta    = (const float*)d_in[5];
    const float* shfA    = (const float*)d_in[6];  // 4
    const float* shfZ    = (const float*)d_in[7];  // 8

    const int total_atoms = in_sizes[0];           // 512
    const int blocks = total_atoms / WAVES;        // 64 blocks, 1 wave per atom
    aev_wmma_kernel<<<blocks, 32 * WAVES, 0, stream>>>(
        species, coords, etaR, shfR, etaA, zeta, shfA, shfZ, (float*)d_out);
}
